// HadamardNorm_9414568313473
// MI455X (gfx1250) — matrix operand order repaired and checked by reading, not compiled, not run on hardware
//
#include <hip/hip_runtime.h>
#include <hip/hip_bf16.h>

#define WAVES 2
#define ROWS  4
#define XPITCH 68            // f32 pitch: 272B rows, 16B aligned, conflict-free
#define TPITCH 72            // bf16 pitch: 144B rows, 16B aligned, conflict-free
#define NEG_SLOPE 0.1f
#define SCALE 0.015625f      // 1/sqrt(4096)

typedef __attribute__((ext_vector_type(16))) __bf16 v16bf;
typedef __attribute__((ext_vector_type(8)))  __bf16 v8bf;
typedef __attribute__((ext_vector_type(8)))  float  v8f;
typedef __attribute__((ext_vector_type(4)))  float  v4f;

union bfbits { unsigned short u; __bf16 b; };

// Packed Dekker split of a float pair:
// hp = {bf16_trunc(x1), bf16_trunc(x0)} via one v_perm_b32;
// lp = {bf16_trunc(x1-h1), bf16_trunc(x0-h0)} (residuals exact in f32).
__device__ __forceinline__ void split_pair(float x0, float x1, unsigned& hp, unsigned& lp) {
  union { float f; unsigned u; } a0, a1, h0, h1, b0, b1;
  a0.f = x0; a1.f = x1;
  hp = __builtin_amdgcn_perm(a1.u, a0.u, 0x07060302u);   // high16(x1) | high16(x0)
  h0.u = a0.u & 0xFFFF0000u;
  h1.u = a1.u & 0xFFFF0000u;
  b0.f = x0 - h0.f;                                      // pk_add (neg) pair
  b1.f = x1 - h1.f;
  lp = __builtin_amdgcn_perm(b1.u, b0.u, 0x07060302u);
}

// Build bf16 hi/lo A-fragment halves from 4 x v4f (fragment elements 0..15 in argument order).
__device__ __forceinline__ void frag_from_f32(v4f pa, v4f pb, v4f pc, v4f pd,
                                              v16bf& fh, v16bf& fl) {
  union { v16bf v; unsigned u[8]; } H, L;
  split_pair(pa[0], pa[1], H.u[0], L.u[0]);
  split_pair(pa[2], pa[3], H.u[1], L.u[1]);
  split_pair(pb[0], pb[1], H.u[2], L.u[2]);
  split_pair(pb[2], pb[3], H.u[3], L.u[3]);
  split_pair(pc[0], pc[1], H.u[4], L.u[4]);
  split_pair(pc[2], pc[3], H.u[5], L.u[5]);
  split_pair(pd[0], pd[1], H.u[6], L.u[6]);
  split_pair(pd[2], pd[3], H.u[7], L.u[7]);
  fh = H.v; fl = L.v;
}

// Low 32 bits of a generic pointer into LDS == LDS byte offset.
__device__ __forceinline__ unsigned lds_off(const void* p) {
  return (unsigned)(unsigned long long)p;
}

// Async DMA one 4096-f32 row into LDS X[64][XPITCH]; 32 x 512B coalesced transfers.
__device__ __forceinline__ void async_copy_row(const float* __restrict__ src,
                                               float* xbuf, int lane) {
#pragma unroll
  for (int c = 0; c < 32; ++c) {
    const int idx4 = c * 32 + lane;
    const int j1 = idx4 >> 4;
    const int j0 = (idx4 & 15) << 2;
    unsigned la = lds_off(xbuf + j1 * XPITCH + j0);
    unsigned long long ga = (unsigned long long)(src + idx4 * 4);
    asm volatile("global_load_async_to_lds_b128 %0, %1, off"
                 :: "v"(la), "v"(ga) : "memory");
  }
}

__device__ __forceinline__ void wait_async0() {
  asm volatile("s_wait_asynccnt 0x0" ::: "memory");
}

// A fragment (16x32 bf16 ISA layout) from bf16 LDS (pitch TPITCH): two ds_load_b128.
__device__ __forceinline__ v16bf a_frag_bf(const __bf16* buf, int mrow, int kcol, int lane) {
  const int m = mrow + (lane & 15);
  const int g = (lane >> 4) << 3;
  union { v16bf v; v8bf h[2]; } u;
  u.h[0] = *(const v8bf*)(buf + m * TPITCH + kcol + g);
  u.h[1] = *(const v8bf*)(buf + m * TPITCH + kcol + 16 + g);
  return u.v;
}

// B fragment (32x16 bf16) of H64 synthesized in registers: H[k][n] = (-1)^popc(k&n).
__device__ __forceinline__ v16bf h_bfrag(int kblk, int nblk, int lane) {
  const int kbase = kblk * 32 + ((lane >> 4) << 3);
  const int n = nblk * 16 + (lane & 15);
  bfbits pos; pos.u = 0x3F80u;
  bfbits neg; neg.u = 0xBF80u;
  v16bf r;
#pragma unroll
  for (int e = 0; e < 16; ++e)
    r[e] = (__popc((kbase + e + (e >= 8 ? 8 : 0)) & n) & 1) ? neg.b : pos.b;
  return r;
}

__device__ __forceinline__ v8f wmma_bf16(v16bf a, v16bf b, v8f c) {
  return __builtin_amdgcn_wmma_f32_16x16x32_bf16(false, a, false, b, (short)0, c, false, false);
}

__global__ __launch_bounds__(32 * WAVES)
void hadamard4096_kernel(const float* __restrict__ in, float* __restrict__ out, int nrows) {
  // per wave: X double buffer (2 x 17408B f32) + T^T hi/lo (2 x 9216B bf16) = 53248B
  __shared__ __attribute__((aligned(16))) unsigned char smem_raw[WAVES * 53248];
  const int lane = threadIdx.x & 31;
  const int w    = threadIdx.x >> 5;
  float*  x0  = (float*)(smem_raw + w * 53248);
  float*  x1  = x0 + 64 * XPITCH;
  __bf16* thi = (__bf16*)(x1 + 64 * XPITCH);
  __bf16* tlo = thi + 64 * TPITCH;

  const long base_row = (long)(blockIdx.x * WAVES + w) * ROWS;

  // Hoisted H B-fragments: reused across both stages and all ROWS rows.
  v16bf B0[4], B1[4];
#pragma unroll
  for (int j = 0; j < 4; ++j) { B0[j] = h_bfrag(0, j, lane); B1[j] = h_bfrag(1, j, lane); }

  if (base_row < nrows) async_copy_row(in + base_row * 4096, x0, lane);

#pragma unroll 1
  for (int rr = 0; rr < ROWS; ++rr) {
    const long row = base_row + rr;
    if (row >= nrows) break;                   // wave-uniform
    float* xb = (rr & 1) ? x1 : x0;

    wait_async0();                             // row rr resident in xb
    if (rr + 1 < ROWS && row + 1 < nrows)      // pipeline: DMA next row during compute
      async_copy_row(in + (row + 1) * 4096, (rr & 1) ? x0 : x1, lane);

    // ---- Stage 1: T = X*H ; store T^T as bf16 hi/lo ----
#pragma unroll
    for (int ti = 0; ti < 4; ++ti) {
      const float* xr = xb + (ti * 16 + (lane & 15)) * XPITCH + ((lane >> 4) << 3);
      v4f r0a = *(const v4f*)(xr + 0),  r0b = *(const v4f*)(xr + 4);
      v4f r1a = *(const v4f*)(xr + 16), r1b = *(const v4f*)(xr + 20);
      v4f r2a = *(const v4f*)(xr + 32), r2b = *(const v4f*)(xr + 36);
      v4f r3a = *(const v4f*)(xr + 48), r3b = *(const v4f*)(xr + 52);
      v16bf a0h, a0l, a1h, a1l;
      frag_from_f32(r0a, r0b, r1a, r1b, a0h, a0l);
      frag_from_f32(r2a, r2b, r3a, r3b, a1h, a1l);
#pragma unroll
      for (int tj = 0; tj < 4; ++tj) {
        v8f acc = {0.f, 0.f, 0.f, 0.f, 0.f, 0.f, 0.f, 0.f};
        acc = wmma_bf16(a0h, B0[tj], acc);
        acc = wmma_bf16(a1h, B1[tj], acc);
        acc = wmma_bf16(a0l, B0[tj], acc);
        acc = wmma_bf16(a1l, B1[tj], acc);
        const int trow = tj * 16 + (lane & 15);
        const int tcol = ti * 16 + ((lane >> 4) << 3);
        union { v8bf v; unsigned u[4]; } H, L;
        split_pair(acc[0], acc[1], H.u[0], L.u[0]);
        split_pair(acc[2], acc[3], H.u[1], L.u[1]);
        split_pair(acc[4], acc[5], H.u[2], L.u[2]);
        split_pair(acc[6], acc[7], H.u[3], L.u[3]);
        *(v8bf*)(thi + trow * TPITCH + tcol) = H.v;
        *(v8bf*)(tlo + trow * TPITCH + tcol) = L.v;
      }
    }

    // ---- Stage 2: U = T^T*H = H X^T H ; y = leaky(U^T/64), stored transposed ----
    float* __restrict__ dst = out + row * 4096;
#pragma unroll
    for (int ti = 0; ti < 4; ++ti) {
      const v16bf a0h = a_frag_bf(thi, ti * 16, 0,  lane);
      const v16bf a0l = a_frag_bf(tlo, ti * 16, 0,  lane);
      const v16bf a1h = a_frag_bf(thi, ti * 16, 32, lane);
      const v16bf a1l = a_frag_bf(tlo, ti * 16, 32, lane);
#pragma unroll
      for (int tj = 0; tj < 4; ++tj) {
        v8f acc = {0.f, 0.f, 0.f, 0.f, 0.f, 0.f, 0.f, 0.f};
        acc = wmma_bf16(a0h, B0[tj], acc);
        acc = wmma_bf16(a1h, B1[tj], acc);
        acc = wmma_bf16(a0l, B0[tj], acc);
        acc = wmma_bf16(a1l, B1[tj], acc);
        const int yrow = tj * 16 + (lane & 15);
        const int ycol = ti * 16 + ((lane >> 4) << 3);
        v4f o0, o1;
#pragma unroll
        for (int r = 0; r < 8; ++r) {
          float a = acc[r] * SCALE;           // positive scale commutes with leaky
          float b = a * NEG_SLOPE;
          float v = fmaxf(a, b);              // leaky(a) = max(a, 0.1a)
          if (r < 4) o0[r] = v; else o1[r - 4] = v;
        }
        *(v4f*)(dst + yrow * 64 + ycol)     = o0;
        *(v4f*)(dst + yrow * 64 + ycol + 4) = o1;
      }
    }
  }
}

extern "C" void kernel_launch(void* const* d_in, const int* in_sizes, int n_in,
                              void* d_out, int out_size, void* d_ws, size_t ws_size,
                              hipStream_t stream) {
  const float* x = (const float*)d_in[0];
  float* y = (float*)d_out;
  const int nrows = in_sizes[0] / 4096;                      // 16384
  const int blocks = (nrows + WAVES * ROWS - 1) / (WAVES * ROWS);
  hipLaunchKernelGGL(hadamard4096_kernel, dim3(blocks), dim3(32 * WAVES), 0, stream,
                     x, y, nrows);
}
